// CrossAttention_43224550867737
// MI455X (gfx1250) — matrix operand order repaired and checked by reading, not compiled, not run on hardware
//
#include <hip/hip_runtime.h>

typedef unsigned short u16;
typedef __attribute__((ext_vector_type(16))) __bf16 v16bf;
typedef __attribute__((ext_vector_type(8)))  float  v8f;

union Frag {
  v16bf bf;
  uint4 u[2];
};

__device__ __forceinline__ u16 f2bf(float f) {
  unsigned u = __float_as_uint(f);
  u += 0x7fffu + ((u >> 16) & 1u);   // round-to-nearest-even
  return (u16)(u >> 16);
}

// ---------------- f32 -> bf16 convert ----------------
__global__ __launch_bounds__(256) void conv_bf16(const float* __restrict__ in,
                                                 u16* __restrict__ out, int n) {
  int i = blockIdx.x * 256 + threadIdx.x;
  if (i < n) out[i] = f2bf(in[i]);
}

// ---------------- transpose+convert weight: Wt[n][k] = W[k][n] * scale ----------------
__global__ __launch_bounds__(256) void transpose_w(const float* __restrict__ W,
                                                   u16* __restrict__ Wt, float scale) {
  __shared__ float tile[16][17];
  int tx = threadIdx.x, ty = threadIdx.y;
  int bx = blockIdx.x * 16, by = blockIdx.y * 16;     // bx: n, by: k
  tile[ty][tx] = W[(size_t)(by + ty) * 1024 + (bx + tx)];
  __syncthreads();
  Wt[(size_t)(bx + ty) * 1024 + (by + tx)] = f2bf(tile[tx][ty] * scale);
}

// ---------------- generic WMMA GEMM ----------------
// C[M,N] = A[M,K](bf16) @ Wt[N,K](bf16, pre-transposed) + bias*biasScale
// Wave tile 64x32 (8 WMMA / k-step). Block: 8 waves = 2(M) x 4(N) -> 128x128.
// mode 0: f32 row-major [M,N]
// mode 1: bf16 head-split  idx = ((b*16+h)*1024 + q)*64 + d      (Q, K)
// mode 2: bf16 head-transposed idx = ((b*16+h)*64 + d)*1024 + q  (V)
__global__ __launch_bounds__(256)
void gemm_wmma(const u16* __restrict__ A, const u16* __restrict__ Wt,
               const float* __restrict__ bias, float biasScale,
               void* __restrict__ Cout, int mode, int M, int N, int K) {
  int lane = threadIdx.x & 31, wave = threadIdx.x >> 5;
  int l16 = lane & 15, hi = lane >> 4;
  int hk8 = hi * 8, hk16 = hi * 16, hr8 = hi * 8;
  int m0 = blockIdx.y * 128 + (wave >> 2) * 64;
  int n0 = blockIdx.x * 128 + (wave & 3) * 32;

  const u16* arow[4];
  const u16* brow[2];
#pragma unroll
  for (int i = 0; i < 4; ++i) arow[i] = A + (size_t)(m0 + i * 16 + l16) * K + hk8;
#pragma unroll
  for (int j = 0; j < 2; ++j) brow[j] = Wt + (size_t)(n0 + j * 16 + l16) * K + hk8;

  v8f acc[4][2];
#pragma unroll
  for (int i = 0; i < 4; ++i)
#pragma unroll
    for (int j = 0; j < 2; ++j) acc[i][j] = (v8f)0.0f;

  for (int k0 = 0; k0 < K; k0 += 32) {
    Frag a[4], b[2];
#pragma unroll
    for (int i = 0; i < 4; ++i) {
      const u16* p = arow[i] + k0;
      a[i].u[0] = *(const uint4*)p;
      a[i].u[1] = *(const uint4*)(p + 16);
    }
#pragma unroll
    for (int j = 0; j < 2; ++j) {
      const u16* p = brow[j] + k0;
      b[j].u[0] = *(const uint4*)p;
      b[j].u[1] = *(const uint4*)(p + 16);
    }
    if (k0 + 128 < K) {                       // keep L2->WGP pipe warm
      __builtin_prefetch(arow[0] + k0 + 128, 0, 1);   // -> global_prefetch_b8
      __builtin_prefetch(brow[0] + k0 + 128, 0, 1);
    }
#pragma unroll
    for (int i = 0; i < 4; ++i)
#pragma unroll
      for (int j = 0; j < 2; ++j)
        acc[i][j] = __builtin_amdgcn_wmma_f32_16x16x32_bf16(
            false, a[i].bf, false, b[j].bf, (short)0, acc[i][j], false, false);
  }

#pragma unroll
  for (int i = 0; i < 4; ++i)
#pragma unroll
    for (int j = 0; j < 2; ++j)
#pragma unroll
      for (int r = 0; r < 8; ++r) {
        int m = m0 + i * 16 + r + hr8;
        int n = n0 + j * 16 + l16;
        float v = acc[i][j][r] + bias[n] * biasScale;
        if (mode == 0) {
          ((float*)Cout)[(size_t)m * N + n] = v;
        } else {
          int bb = m >> 10, q = m & 1023, h = n >> 6, d = n & 63;
          u16 bv = f2bf(v);
          if (mode == 1)
            ((u16*)Cout)[(((size_t)bb * 16 + h) * 1024 + q) * 64 + d] = bv;
          else
            ((u16*)Cout)[(((size_t)bb * 16 + h) * 64 + d) * 1024 + q] = bv;
        }
      }
}

// ---------------- flash attention ----------------
// Qb,Kb: bf16 [B,H,1024,64]; Vt: bf16 [B,H,64,1024]; Out: bf16 [B,1024,1024]
__global__ __launch_bounds__(256)
void attn_wmma(const u16* __restrict__ Qb, const u16* __restrict__ Kb,
               const u16* __restrict__ Vt, u16* __restrict__ Out) {
  __shared__ __align__(16) u16 sP[8 * 16 * 64];  // per-wave 16x64 P tile, bf16
  int lane = threadIdx.x & 31, wave = threadIdx.x >> 5;
  int l16 = lane & 15, hi = lane >> 4;
  int hk8 = hi * 8, hk16 = hi * 16, hr8 = hi * 8;
  int b = blockIdx.z, h = blockIdx.y;
  int bh = b * 16 + h;
  int q0 = blockIdx.x * 128 + wave * 16;
  const u16* Qbase = Qb + (size_t)bh * 1024 * 64;
  const u16* Kbase = Kb + (size_t)bh * 1024 * 64;
  const u16* Vbase = Vt + (size_t)bh * 64 * 1024;
  int waveBase = wave * 16 * 64;

  // Q fragments for this wave's 16 rows (softmax scale folded into Wq/bq)
  Frag qf[2];
#pragma unroll
  for (int t = 0; t < 2; ++t) {
    const u16* p = Qbase + (size_t)(q0 + l16) * 64 + t * 32 + hk8;
    qf[t].u[0] = *(const uint4*)p;
    qf[t].u[1] = *(const uint4*)(p + 16);
  }

  v8f O[4];
#pragma unroll
  for (int t = 0; t < 4; ++t) O[t] = (v8f)0.0f;
  float mrow[8], lrow[8];
#pragma unroll
  for (int r = 0; r < 8; ++r) { mrow[r] = -1e30f; lrow[r] = 0.0f; }

  for (int kc = 0; kc < 1024; kc += 64) {
    if (kc + 64 < 1024) {
      __builtin_prefetch(Kbase + (size_t)(kc + 64 + l16) * 64, 0, 1);
      __builtin_prefetch(Vbase + (size_t)l16 * 1024 + kc + 64, 0, 1);
    }
    // ---- scores S = Q @ K^T, 16 x 64 chunk ----
    v8f S[4];
#pragma unroll
    for (int j = 0; j < 4; ++j) {
      S[j] = (v8f)0.0f;
#pragma unroll
      for (int t = 0; t < 2; ++t) {
        Frag kf;
        const u16* p = Kbase + (size_t)(kc + j * 16 + l16) * 64 + t * 32 + hk8;
        kf.u[0] = *(const uint4*)p;
        kf.u[1] = *(const uint4*)(p + 16);
        S[j] = __builtin_amdgcn_wmma_f32_16x16x32_bf16(
            false, qf[t].bf, false, kf.bf, (short)0, S[j], false, false);
      }
    }
    // ---- online softmax (rows split across half-waves) ----
    float mn[8];
#pragma unroll
    for (int r = 0; r < 8; ++r) {
      mn[r] = mrow[r];
#pragma unroll
      for (int j = 0; j < 4; ++j) mn[r] = fmaxf(mn[r], S[j][r]);
#pragma unroll
      for (int off = 1; off <= 8; off <<= 1)
        mn[r] = fmaxf(mn[r], __shfl_xor(mn[r], off, 32));
    }
    float alpha[8], rs[8];
#pragma unroll
    for (int r = 0; r < 8; ++r) {
      alpha[r] = __expf(mrow[r] - mn[r]);
      mrow[r] = mn[r];
      rs[r] = 0.0f;
    }
#pragma unroll
    for (int t = 0; t < 4; ++t)
#pragma unroll
      for (int r = 0; r < 8; ++r) O[t][r] *= alpha[r];
#pragma unroll
    for (int j = 0; j < 4; ++j)
#pragma unroll
      for (int r = 0; r < 8; ++r) {
        float p = __expf(S[j][r] - mn[r]);
        S[j][r] = p;
        rs[r] += p;
      }
#pragma unroll
    for (int r = 0; r < 8; ++r) {
#pragma unroll
      for (int off = 1; off <= 8; off <<= 1) rs[r] += __shfl_xor(rs[r], off, 32);
      lrow[r] = lrow[r] * alpha[r] + rs[r];
    }
    // ---- P (C-layout) -> LDS -> A-layout fragments ----
    __syncthreads();
#pragma unroll
    for (int j = 0; j < 4; ++j)
#pragma unroll
      for (int r = 0; r < 8; ++r)
        sP[waveBase + (r + hr8) * 64 + j * 16 + l16] = f2bf(S[j][r]);
    __syncthreads();
    Frag pf[2];
#pragma unroll
    for (int j2 = 0; j2 < 2; ++j2) {
      const u16* p = &sP[waveBase + l16 * 64 + j2 * 32 + hk8];
      pf[j2].u[0] = *(const uint4*)p;
      pf[j2].u[1] = *(const uint4*)(p + 16);
    }
    // ---- O += P @ V ----
#pragma unroll
    for (int t = 0; t < 4; ++t)
#pragma unroll
      for (int j2 = 0; j2 < 2; ++j2) {
        Frag vf;
        const u16* vp = Vbase + (size_t)(t * 16 + l16) * 1024 + kc + j2 * 32 + hk8;
        vf.u[0] = *(const uint4*)vp;
        vf.u[1] = *(const uint4*)(vp + 16);
        O[t] = __builtin_amdgcn_wmma_f32_16x16x32_bf16(
            false, pf[j2].bf, false, vf.bf, (short)0, O[t], false, false);
      }
  }

  // ---- normalize and store merged-head bf16 output [B,1024,1024] ----
  float inv[8];
#pragma unroll
  for (int r = 0; r < 8; ++r) inv[r] = 1.0f / lrow[r];
#pragma unroll
  for (int t = 0; t < 4; ++t)
#pragma unroll
    for (int r = 0; r < 8; ++r) {
      float v = O[t][r] * inv[r];
      Out[((size_t)b * 1024 + q0 + r + hr8) * 1024 + h * 64 + t * 16 + l16] = f2bf(v);
    }
}

extern "C" void kernel_launch(void* const* d_in, const int* in_sizes, int n_in,
                              void* d_out, int out_size, void* d_ws, size_t ws_size,
                              hipStream_t stream) {
  const float* x   = (const float*)d_in[0];
  const float* ctx = (const float*)d_in[1];
  const float* Wq  = (const float*)d_in[2];
  const float* bq  = (const float*)d_in[3];
  const float* Wk  = (const float*)d_in[4];
  const float* bk  = (const float*)d_in[5];
  const float* Wv  = (const float*)d_in[6];
  const float* bv  = (const float*)d_in[7];
  const float* Wo  = (const float*)d_in[8];
  const float* bo  = (const float*)d_in[9];
  float* out = (float*)d_out;

  const int M = 8 * 1024, N = 1024, K = 1024;
  const int nAct = 8 * 1024 * 1024;
  const size_t ACT  = (size_t)nAct * 2;       // 16 MiB bf16 activation
  const size_t WMAT = (size_t)1024 * 1024 * 2;

  char* ws = (char*)d_ws;
  u16* xb  = (u16*)ws; ws += ACT;   // x bf16; reused later as attention output
  u16* cb  = (u16*)ws; ws += ACT;   // context bf16
  u16* Qb  = (u16*)ws; ws += ACT;   // [B,H,Nq,64]
  u16* Kb  = (u16*)ws; ws += ACT;   // [B,H,Nk,64]
  u16* Vt  = (u16*)ws; ws += ACT;   // [B,H,64,Nk]
  u16* Wqt = (u16*)ws; ws += WMAT;
  u16* Wkt = (u16*)ws; ws += WMAT;
  u16* Wvt = (u16*)ws; ws += WMAT;
  u16* Wot = (u16*)ws; ws += WMAT;

  const float qscale = 0.125f;      // 1/sqrt(64), folded into Wq & bq

  conv_bf16<<<nAct / 256, 256, 0, stream>>>(x, xb, nAct);
  conv_bf16<<<nAct / 256, 256, 0, stream>>>(ctx, cb, nAct);

  dim3 tb(16, 16), tg(64, 64);
  transpose_w<<<tg, tb, 0, stream>>>(Wq, Wqt, qscale);
  transpose_w<<<tg, tb, 0, stream>>>(Wk, Wkt, 1.0f);
  transpose_w<<<tg, tb, 0, stream>>>(Wv, Wvt, 1.0f);
  transpose_w<<<tg, tb, 0, stream>>>(Wo, Wot, 1.0f);

  dim3 gg(N / 128, M / 128);
  gemm_wmma<<<gg, 256, 0, stream>>>(xb, Wqt, bq, qscale, Qb, 1, M, N, K);
  gemm_wmma<<<gg, 256, 0, stream>>>(cb, Wkt, bk, 1.0f,   Kb, 1, M, N, K);
  gemm_wmma<<<gg, 256, 0, stream>>>(cb, Wvt, bv, 1.0f,   Vt, 2, M, N, K);

  attn_wmma<<<dim3(8, 16, 8), 256, 0, stream>>>(Qb, Kb, Vt, xb);  // xb reused as attn out

  gemm_wmma<<<gg, 256, 0, stream>>>(xb, Wot, bo, 1.0f, out, 0, M, N, K);
}
